// GroupedQueryAttention_33981781246587
// MI455X (gfx1250) — matrix operand order repaired and checked by reading, not compiled, not run on hardware
//
#include <hip/hip_runtime.h>
#include <cmath>
#include <stdint.h>

typedef __attribute__((ext_vector_type(16))) __bf16 v16bf;
typedef __attribute__((ext_vector_type(8)))  __bf16 v8bf;
typedef __attribute__((ext_vector_type(8)))  float  v8f;

#define B_    4
#define S_    1024
#define HID_  2048
#define NH_   16
#define NKV_  4
#define HD_   128
#define SCALE_ 0.08838834764831845f   // 1/sqrt(128)

// ---------- fragment helpers (CDNA5 wave32 WMMA layouts) ----------

static __device__ __forceinline__ v16bf ld_pair8(const __bf16* lo, const __bf16* hi) {
  v8bf a = *(const v8bf*)lo;
  v8bf b = *(const v8bf*)hi;
  v16bf r;
#pragma unroll
  for (int j = 0; j < 8; ++j) { r[j] = a[j]; r[j + 8] = b[j]; }
  return r;
}
static __device__ __forceinline__ v16bf ld_cont16(const __bf16* p) {
  return ld_pair8(p, p + 16);
}

static __device__ __forceinline__ v8f wmma_bf16(v16bf a, v16bf b, v8f c) {
  return __builtin_amdgcn_wmma_f32_16x16x32_bf16(false, a, false, b, (short)0, c,
                                                 false, false);
}

static __device__ __forceinline__ float rowmax16(float v) {
#pragma unroll
  for (int m = 1; m < 16; m <<= 1) v = fmaxf(v, __shfl_xor(v, m, 32));
  return v;
}
static __device__ __forceinline__ float rowsum16(float v) {
#pragma unroll
  for (int m = 1; m < 16; m <<= 1) v += __shfl_xor(v, m, 32);
  return v;
}

// ---------- CDNA5 async global->LDS copy (ASYNCcnt-tracked) ----------

static __device__ __forceinline__ void async_copy_b128(uint32_t lds_addr,
                                                       const void* gaddr) {
  asm volatile("global_load_async_to_lds_b128 %0, %1, off"
               :
               : "v"(lds_addr), "v"((uint64_t)(uintptr_t)gaddr)
               : "memory");
}
static __device__ __forceinline__ void wait_async0() {
  asm volatile("s_wait_asynccnt 0" ::: "memory");
}

// ---------- kernel 1: Y = X @ W, f32 in, bf16 out in [B, heads, S, D] ----------
// 128x64 block tile, 8 waves, each wave owns a 32x32 sub-tile (4 WMMA / k-step).

__global__ __launch_bounds__(256)
void gqa_gemm_qkv(const float* __restrict__ X, const float* __restrict__ W,
                  __bf16* __restrict__ Out, int N) {
  __shared__ __align__(16) __bf16 As[128][40];  // 128 rows x 32 K (+8 pad)
  __shared__ __align__(16) __bf16 BsT[64][40];  // 64 cols x 32 K (+8 pad)

  const int tid = threadIdx.x;
  const int lane = tid & 31, wave = tid >> 5;
  const int wm = wave >> 1, wn = wave & 1;     // 4x2 waves of 32x32
  const int half = lane >> 4, l16 = lane & 15;
  const int m0 = blockIdx.y * 128, n0 = blockIdx.x * 64;

  const int arow = tid >> 1, acol = (tid & 1) * 16;  // 16 f32 per thread
  const int bk   = tid >> 3, bn   = (tid & 7) * 8;
  const float* srcA = X + (size_t)(m0 + arow) * HID_ + acol;
  const float* srcB = W + (size_t)bk * N + n0 + bn;

  v8f c00 = {}, c01 = {}, c10 = {}, c11 = {};
  for (int k0 = 0; k0 < HID_; k0 += 32) {
#pragma unroll
    for (int j = 0; j < 16; ++j) As[arow][acol + j] = (__bf16)srcA[k0 + j];
    const float* pb = srcB + (size_t)k0 * N;
#pragma unroll
    for (int j = 0; j < 8; ++j) BsT[bn + j][bk] = (__bf16)pb[j];
    if (k0 + 32 < HID_) {
      __builtin_prefetch(srcA + k0 + 32, 0, 0);                // global_prefetch_b8
      __builtin_prefetch(srcB + (size_t)(k0 + 32) * N, 0, 0);
    }
    __syncthreads();

    v16bf a0 = ld_pair8(&As[wm * 32 + l16][half * 8],
                        &As[wm * 32 + l16][half * 8 + 16]);
    v16bf a1 = ld_pair8(&As[wm * 32 + 16 + l16][half * 8],
                        &As[wm * 32 + 16 + l16][half * 8 + 16]);
    v16bf b0 = ld_cont16(&BsT[wn * 32 + l16][half * 8]);
    v16bf b1 = ld_cont16(&BsT[wn * 32 + 16 + l16][half * 8]);
    c00 = wmma_bf16(a0, b0, c00);
    c01 = wmma_bf16(a0, b1, c01);
    c10 = wmma_bf16(a1, b0, c10);
    c11 = wmma_bf16(a1, b1, c11);
    __syncthreads();
  }

  const int heads = N >> 7;  // N / HEAD_DIM
  v8f acc[2][2] = {{c00, c01}, {c10, c11}};
#pragma unroll
  for (int am = 0; am < 2; ++am) {
#pragma unroll
    for (int bn_ = 0; bn_ < 2; ++bn_) {
#pragma unroll
      for (int i = 0; i < 8; ++i) {
        int row = m0 + wm * 32 + am * 16 + half * 8 + i;
        int col = n0 + wn * 32 + bn_ * 16 + l16;
        int bi = row >> 10, s = row & (S_ - 1);
        int h = col >> 7, d = col & (HD_ - 1);
        Out[(((size_t)bi * heads + h) * S_ + s) * HD_ + d] = (__bf16)acc[am][bn_][i];
      }
    }
  }
}

// ---------- kernel 2: in-place RoPE on bf16 [B, heads, S, D] ----------

__global__ __launch_bounds__(256)
void gqa_rope(__bf16* __restrict__ Xh, int heads) {
  int idx = blockIdx.x * blockDim.x + threadIdx.x;
  int total = B_ * heads * S_ * 64;
  if (idx >= total) return;
  int i  = idx & 63;                 // rotary pair index 0..63
  int s  = (idx >> 6) & (S_ - 1);    // position
  int bh = idx >> 16;                // (b*heads + h) since S_*64 == 1<<16
  __bf16* row = Xh + ((size_t)bh * S_ + s) * HD_;
  float x0 = (float)row[i];
  float x1 = (float)row[i + 64];
  float inv = powf(10000.0f, -(float)(2 * i) * (1.0f / 128.0f));
  float ang = (float)s * inv;
  float sv, cv;
  sincosf(ang, &sv, &cv);
  row[i]      = (__bf16)(x0 * cv - x1 * sv);
  row[i + 64] = (__bf16)(x1 * cv + x0 * sv);
}

// ---------- kernel 3: causal GQA flash attention, bf16 in/out ----------

__global__ __launch_bounds__(128)
void gqa_attn(const __bf16* __restrict__ Qb, const __bf16* __restrict__ Kb,
              const __bf16* __restrict__ Vb, __bf16* __restrict__ Ctx) {
  __shared__ __align__(16) __bf16 Ks[32][136];   // 32 keys x 128 d (+8 pad)
  __shared__ __align__(16) __bf16 VsT[128][40];  // 128 d x 32 keys (+8 pad)
  __shared__ __align__(16) __bf16 Ps[4][16][40]; // per-wave P tile round-trip

  const int tid = threadIdx.x;
  const int lane = tid & 31, w = tid >> 5;
  const int half = lane >> 4, l16 = lane & 15;
  const int q0 = blockIdx.x * 64;
  const int h  = blockIdx.y;
  const int bi = blockIdx.z;
  const int kvh = h >> 2;  // G = NUM_HEADS / NUM_KV_HEADS = 4

  const __bf16* Khead = Kb + ((size_t)bi * NKV_ + kvh) * S_ * HD_;
  const __bf16* Vhead = Vb + ((size_t)bi * NKV_ + kvh) * S_ * HD_;

  // Q fragments for this wave's 16 rows, all 4 K-chunks, loaded once.
  const __bf16* qrow =
      Qb + (((size_t)bi * NH_ + h) * S_ + q0 + w * 16 + l16) * HD_;
  v16bf qf[4];
#pragma unroll
  for (int c = 0; c < 4; ++c)
    qf[c] = ld_pair8(qrow + 32 * c + half * 8, qrow + 32 * c + half * 8 + 16);

  v8f zero = {};
  v8f acc[8];
#pragma unroll
  for (int dt = 0; dt < 8; ++dt) acc[dt] = zero;
  float m[8], l[8];
#pragma unroll
  for (int i = 0; i < 8; ++i) { m[i] = -INFINITY; l[i] = 0.0f; }

  const int sk = tid >> 2, sd = (tid & 3) * 32;  // staging: 32 elems/thread
  const uint32_t ldsK = (uint32_t)(uintptr_t)(&Ks[sk][sd]);

  for (int kt = 0; kt < q0 + 64; kt += 32) {
    __syncthreads();
    const __bf16* ksrc = Khead + (size_t)(kt + sk) * HD_ + sd;
    const __bf16* vsrc = Vhead + (size_t)(kt + sk) * HD_ + sd;
    // K tile: async global->LDS DMA, 64 contiguous bytes per lane
#pragma unroll
    for (int u = 0; u < 4; ++u)
      async_copy_b128(ldsK + 16 * u, ksrc + 8 * u);
    // V tile: transpose through VGPRs
#pragma unroll
    for (int j = 0; j < 32; ++j) VsT[sd + j][sk] = vsrc[j];
    wait_async0();
    __syncthreads();

    // scores: two 16x16 tiles covering keys kt..kt+31
    v8f s0 = zero, s1 = zero;
#pragma unroll
    for (int c = 0; c < 4; ++c) {
      v16bf b0 = ld_cont16(&Ks[l16][32 * c + half * 8]);
      v16bf b1 = ld_cont16(&Ks[16 + l16][32 * c + half * 8]);
      s0 = wmma_bf16(qf[c], b0, s0);
      s1 = wmma_bf16(qf[c], b1, s1);
    }

    const int col0 = kt + l16, col1 = col0 + 16;
    float corr[8];
#pragma unroll
    for (int i = 0; i < 8; ++i) {
      const int rowq = q0 + w * 16 + half * 8 + i;
      float a0 = (col0 <= rowq) ? s0[i] * SCALE_ : -INFINITY;
      float a1 = (col1 <= rowq) ? s1[i] * SCALE_ : -INFINITY;
      float mn = fmaxf(m[i], rowmax16(fmaxf(a0, a1)));
      float cr = expf(m[i] - mn);
      float p0 = expf(a0 - mn);
      float p1 = expf(a1 - mn);
      l[i] = l[i] * cr + rowsum16(p0 + p1);
      m[i] = mn;
      corr[i] = cr;
      Ps[w][half * 8 + i][l16]      = (__bf16)p0;
      Ps[w][half * 8 + i][16 + l16] = (__bf16)p1;
    }
#pragma unroll
    for (int dt = 0; dt < 8; ++dt)
#pragma unroll
      for (int i = 0; i < 8; ++i) acc[dt][i] *= corr[i];

    // reload P in A-fragment layout (intra-wave LDS round trip)
    v16bf pa = ld_pair8(&Ps[w][l16][half * 8], &Ps[w][l16][half * 8 + 16]);
#pragma unroll
    for (int dt = 0; dt < 8; ++dt) {
      v16bf bv = ld_cont16(&VsT[dt * 16 + l16][half * 8]);
      acc[dt] = wmma_bf16(pa, bv, acc[dt]);
    }
  }

  // ctx epilogue: [B, S, H*D] bf16 for the out-projection GEMM
#pragma unroll
  for (int dt = 0; dt < 8; ++dt) {
#pragma unroll
    for (int i = 0; i < 8; ++i) {
      int srow = q0 + w * 16 + half * 8 + i;
      float o = acc[dt][i] / l[i];
      Ctx[((size_t)bi * S_ + srow) * HID_ + h * HD_ + dt * 16 + l16] = (__bf16)o;
    }
  }
}

// ---------- kernel 4: out = ctx(bf16) @ Wo(f32), fp32 row-major out ----------

__global__ __launch_bounds__(256)
void gqa_gemm_out(const __bf16* __restrict__ Ctx, const float* __restrict__ Wo,
                  float* __restrict__ Out) {
  __shared__ __align__(16) __bf16 As[128][40];
  __shared__ __align__(16) __bf16 BsT[64][40];

  const int tid = threadIdx.x;
  const int lane = tid & 31, wave = tid >> 5;
  const int wm = wave >> 1, wn = wave & 1;
  const int half = lane >> 4, l16 = lane & 15;
  const int m0 = blockIdx.y * 128, n0 = blockIdx.x * 64;

  const int arow = tid >> 1, acol = (tid & 1) * 16;
  const int bk   = tid >> 3, bn   = (tid & 7) * 8;
  const __bf16* srcA = Ctx + (size_t)(m0 + arow) * HID_ + acol;
  const float*  srcB = Wo + (size_t)bk * HID_ + n0 + bn;

  v8f c00 = {}, c01 = {}, c10 = {}, c11 = {};
  for (int k0 = 0; k0 < HID_; k0 += 32) {
#pragma unroll
    for (int j = 0; j < 16; ++j) As[arow][acol + j] = srcA[k0 + j];
    const float* pb = srcB + (size_t)k0 * HID_;
#pragma unroll
    for (int j = 0; j < 8; ++j) BsT[bn + j][bk] = (__bf16)pb[j];
    if (k0 + 32 < HID_) {
      __builtin_prefetch(srcA + k0 + 32, 0, 0);
      __builtin_prefetch(srcB + (size_t)(k0 + 32) * HID_, 0, 0);
    }
    __syncthreads();

    v16bf a0 = ld_pair8(&As[wm * 32 + l16][half * 8],
                        &As[wm * 32 + l16][half * 8 + 16]);
    v16bf a1 = ld_pair8(&As[wm * 32 + 16 + l16][half * 8],
                        &As[wm * 32 + 16 + l16][half * 8 + 16]);
    v16bf b0 = ld_cont16(&BsT[wn * 32 + l16][half * 8]);
    v16bf b1 = ld_cont16(&BsT[wn * 32 + 16 + l16][half * 8]);
    c00 = wmma_bf16(a0, b0, c00);
    c01 = wmma_bf16(a0, b1, c01);
    c10 = wmma_bf16(a1, b0, c10);
    c11 = wmma_bf16(a1, b1, c11);
    __syncthreads();
  }

  v8f acc[2][2] = {{c00, c01}, {c10, c11}};
#pragma unroll
  for (int am = 0; am < 2; ++am) {
#pragma unroll
    for (int bn_ = 0; bn_ < 2; ++bn_) {
#pragma unroll
      for (int i = 0; i < 8; ++i) {
        int row = m0 + wm * 32 + am * 16 + half * 8 + i;
        int col = n0 + wn * 32 + bn_ * 16 + l16;
        Out[(size_t)row * HID_ + col] = acc[am][bn_][i];
      }
    }
  }
}

// ---------- host ----------

extern "C" void kernel_launch(void* const* d_in, const int* in_sizes, int n_in,
                              void* d_out, int out_size, void* d_ws, size_t ws_size,
                              hipStream_t stream) {
  (void)in_sizes; (void)n_in; (void)out_size; (void)ws_size;

  const float* X  = (const float*)d_in[0];  // [B,S,HID]
  const float* Wq = (const float*)d_in[1];  // [HID, 2048]
  const float* Wk = (const float*)d_in[2];  // [HID, 512]
  const float* Wv = (const float*)d_in[3];  // [HID, 512]
  const float* Wo = (const float*)d_in[4];  // [2048, HID]
  float* Out = (float*)d_out;               // [B,S,HID] f32

  // workspace: Q (16MB) | K (4MB) | V (4MB) | ctx (16MB), all bf16
  __bf16* Qb  = (__bf16*)d_ws;
  __bf16* Kb  = Qb + (size_t)B_ * NH_  * S_ * HD_;
  __bf16* Vb  = Kb + (size_t)B_ * NKV_ * S_ * HD_;
  __bf16* Ctx = Vb + (size_t)B_ * NKV_ * S_ * HD_;

  const int M = B_ * S_;  // 4096

  gqa_gemm_qkv<<<dim3(HID_ / 64, M / 128), 256, 0, stream>>>(X, Wq, Qb, HID_);
  gqa_gemm_qkv<<<dim3(512 / 64,  M / 128), 256, 0, stream>>>(X, Wk, Kb, 512);
  gqa_gemm_qkv<<<dim3(512 / 64,  M / 128), 256, 0, stream>>>(X, Wv, Vb, 512);

  gqa_rope<<<(B_ * NH_  * S_ * 64 + 255) / 256, 256, 0, stream>>>(Qb, NH_);
  gqa_rope<<<(B_ * NKV_ * S_ * 64 + 255) / 256, 256, 0, stream>>>(Kb, NKV_);

  gqa_attn<<<dim3(S_ / 64, NH_, B_), 128, 0, stream>>>(Qb, Kb, Vb, Ctx);

  gqa_gemm_out<<<dim3(HID_ / 64, M / 128), 256, 0, stream>>>(Ctx, Wo, Out);
}
